// DGCNN_70617852281555
// MI455X (gfx1250) — compile-verified, matrix operand order checked by reading, not run on hardware
//
#include <hip/hip_runtime.h>
#include <hip/hip_bf16.h>

// ---------------------------------------------------------------------------
// DGCNN forward for MI455X (gfx1250, wave32, WMMA).
//
// Decomposition: W*[xj-xi; xi] = W1*xj + (W2-W1)*xi  ==> per layer:
//   1) gram X*X^T via v_wmma_f32_16x16x32_f16 fused with LDS top-20 (kNN)
//   2) A = X*W1^T , Bv = X*(W2-W1)^T as one WMMA GEMM into [N, 2O]
//   3) h[n,o] = max_j lrelu(A[idx[n,j],o] + Bv[n,o] + b[o])  (gather kernel)
// Final: [N,512] x [512,1024] WMMA GEMM (weights staged in LDS via
// global_load_async_to_lds_b128, then resident in VGPRs) fused with
// bias/lrelu/max over N.  __launch_bounds__ sized so weight fragments do
// not spill (2 waves/SIMD -> large VGPR budget).
// Workspace ~121 MB (fits in MI455X 192MB L2 -> compute/irregular bound).
// ---------------------------------------------------------------------------

#define BATCH 16
#define NPTS  2048
#define KNN   20
#define CCAT  512

typedef __attribute__((ext_vector_type(16))) _Float16 v16h;
typedef __attribute__((ext_vector_type(8)))  _Float16 v8h;
typedef __attribute__((ext_vector_type(8)))  float    v8f;

// Per-lane fragment loader for 16x32 f16 tiles (A operand; B uses the same
// packing for the transposed operand). Assumed CDNA5 layout (ISA 7.12.2):
// lane<16: row M=lane, K in {0..7, 16..23}; lane>=16: same row, K in {8..15, 24..31}.
__device__ __forceinline__ v16h load_frag(const _Float16* p,
                                          int row_base, int stride, int k0) {
  int lane = threadIdx.x & 31;
  int row  = row_base + (lane & 15);
  int kb   = k0 + ((lane >> 4) << 3);
  const _Float16* q = p + (size_t)row * stride + kb;
  v8h lo = *(const v8h*)(q);        // K = kb .. kb+7
  v8h hi = *(const v8h*)(q + 16);   // K = kb+16 .. kb+23
  return __builtin_shufflevector(lo, hi, 0,1,2,3,4,5,6,7,8,9,10,11,12,13,14,15);
}

__device__ __forceinline__ v8f wmma32(v16h a, v16h b, v8f c) {
  return __builtin_amdgcn_wmma_f32_16x16x32_f16(false, a, false, b,
                                                (short)0, c, false, false);
}

// ---------------- input prep: x[B,N,3] f32 -> h16[B,N,32] f16 + sqnorm -----
__global__ void __launch_bounds__(256)
k_prep_x(const float* __restrict__ x,
         _Float16* __restrict__ h16, float* __restrict__ sq) {
  int i = blockIdx.x * blockDim.x + threadIdx.x;       // b*N + n
  if (i >= BATCH * NPTS) return;
  float a0 = x[i*3+0], a1 = x[i*3+1], a2 = x[i*3+2];
  _Float16* p = h16 + (size_t)i * 32;
  p[0] = (_Float16)a0; p[1] = (_Float16)a1; p[2] = (_Float16)a2;
  #pragma unroll
  for (int c = 3; c < 32; ++c) p[c] = (_Float16)0.f;
  sq[i] = a0*a0 + a1*a1 + a2*a2;
}

// ---------------- weight fold: W[O,2C] f32 -> [W1; W2-W1] f16 [2O, CP] -----
__global__ void __launch_bounds__(128)
k_prep_w(const float* __restrict__ W, _Float16* __restrict__ wo,
         int cin, int cp, int O) {
  int r = blockIdx.x, c = threadIdx.x;                 // r in [0,2O), c in [0,cp)
  float v = 0.f;
  if (c < cin) {
    if (r < O) v = W[(size_t)r * 2 * cin + c];
    else {
      int rr = r - O;
      v = W[(size_t)rr * 2 * cin + cin + c] - W[(size_t)rr * 2 * cin + c];
    }
  }
  wo[(size_t)r * cp + c] = (_Float16)v;
}

__global__ void __launch_bounds__(256)
k_prep_wf(const float* __restrict__ Wf, _Float16* __restrict__ wo) {
  int r = blockIdx.x;
  for (int c = threadIdx.x; c < CCAT; c += blockDim.x)
    wo[(size_t)r * CCAT + c] = (_Float16)Wf[(size_t)r * CCAT + c];
}

// ---------------- kNN: WMMA gram tiles fused with LDS top-20 --------------
template<int NKC>
__global__ void __launch_bounds__(32)
k_knn(const _Float16* __restrict__ H, const float* __restrict__ sq,
      int* __restrict__ idx) {
  const int cp = NKC * 32;
  int b = blockIdx.y, t0 = blockIdx.x * 16;
  const _Float16* hb  = H  + (size_t)b * NPTS * cp;
  const float*    sqb = sq + (size_t)b * NPTS;
  int lane = threadIdx.x & 31;
  int col  = lane & 15, mhi = (lane >> 4) * 8;

  __shared__ float dt[16][16];
  __shared__ float topd[16][KNN];
  __shared__ int   topi[16][KNN];
  if (lane < 16)
    for (int j = 0; j < KNN; ++j) topd[lane][j] = -3.4e38f;

  v16h afr[NKC];
  #pragma unroll
  for (int kc = 0; kc < NKC; ++kc) afr[kc] = load_frag(hb, t0, cp, kc * 32);

  float sqr[8];
  #pragma unroll
  for (int r = 0; r < 8; ++r) sqr[r] = sqb[t0 + r + mhi];

  for (int mt = 0; mt < NPTS / 16; ++mt) {
    int m0 = mt * 16;
    if (mt + 1 < NPTS / 16)
      __builtin_prefetch(hb + (size_t)(m0 + 16 + (lane & 15)) * cp, 0, 1);
    v8f acc = {};
    #pragma unroll
    for (int kc = 0; kc < NKC; ++kc) {
      v16h bfr = load_frag(hb, m0, cp, kc * 32);
      acc = wmma32(afr[kc], bfr, acc);
    }
    float sqc = sqb[m0 + col];
    #pragma unroll
    for (int r = 0; r < 8; ++r)
      dt[r + mhi][col] = 2.f * acc[r] - sqr[r] - sqc;   // neg sq distance
    __syncthreads();
    if (lane < 16) {
      float* td = topd[lane]; int* ti = topi[lane];
      for (int c = 0; c < 16; ++c) {
        float d = dt[lane][c];
        if (d > td[KNN - 1]) {
          int p = KNN - 1;
          while (p > 0 && td[p-1] < d) { td[p] = td[p-1]; ti[p] = ti[p-1]; --p; }
          td[p] = d; ti[p] = m0 + c;
        }
      }
    }
    __syncthreads();
  }
  if (lane < 16) {
    size_t o = ((size_t)b * NPTS + t0 + lane) * KNN;
    for (int j = 0; j < KNN; ++j) idx[o + j] = topi[lane][j];
  }
}

// ------- WMMA GEMM: out[b][N][ocols] = X * W^T, OT o-tiles per wave --------
template<int NKC, int OT>
__global__ void __launch_bounds__(32)
k_gemm16(const _Float16* __restrict__ X, const _Float16* __restrict__ Wt,
         float* __restrict__ outp, int ocols) {
  const int cp = NKC * 32;
  int b  = blockIdx.z;
  int n0 = blockIdx.x * 16;
  int o0 = blockIdx.y * (16 * OT);
  const _Float16* xb = X + (size_t)b * NPTS * cp;
  v8f acc[OT];
  #pragma unroll
  for (int t = 0; t < OT; ++t) acc[t] = (v8f){};
  #pragma unroll
  for (int kc = 0; kc < NKC; ++kc) {
    v16h a = load_frag(xb, n0, cp, kc * 32);       // reused across OT tiles
    #pragma unroll
    for (int t = 0; t < OT; ++t) {
      v16h w = load_frag(Wt, o0 + 16 * t, cp, kc * 32);
      acc[t] = wmma32(a, w, acc[t]);
    }
  }
  int lane = threadIdx.x & 31;
  int col = lane & 15, mhi = (lane >> 4) * 8;
  float* ob = outp + (size_t)b * NPTS * ocols;
  #pragma unroll
  for (int t = 0; t < OT; ++t)
    #pragma unroll
    for (int r = 0; r < 8; ++r)
      ob[(size_t)(n0 + r + mhi) * ocols + o0 + 16 * t + col] = acc[t][r];
}

// ---------------- edge gather + lrelu + max over k -------------------------
__global__ void __launch_bounds__(256)
k_edge_max(const float* __restrict__ ab, const float* __restrict__ bias,
           const int* __restrict__ idx, _Float16* __restrict__ hnext,
           _Float16* __restrict__ cat, float* __restrict__ sqout, int O) {
  int b = blockIdx.y, n = blockIdx.x, o = threadIdx.x;
  size_t pt = (size_t)b * NPTS + n;
  const float* abB = ab + (size_t)b * NPTS * (2 * O);
  float bv = abB[(size_t)n * (2 * O) + O + o] + bias[o];
  const int* ix = idx + pt * KNN;
  float mx = -3.4e38f;
  #pragma unroll 4
  for (int j = 0; j < KNN; ++j) {
    int jn = ix[j];
    float v = abB[(size_t)jn * (2 * O) + o] + bv;
    v = v > 0.f ? v : 0.2f * v;
    mx = fmaxf(mx, v);
  }
  cat[pt * CCAT + o] = (_Float16)mx;                 // cat pre-offset by caller
  if (hnext) hnext[pt * O + o] = (_Float16)mx;
  if (sqout) {
    __shared__ float red[256];
    red[o] = mx * mx;
    __syncthreads();
    for (int s = blockDim.x >> 1; s > 0; s >>= 1) {
      if (o < s) red[o] += red[o + s];
      __syncthreads();
    }
    if (o == 0) sqout[pt] = red[0];
  }
}

// ---------------- final GEMM [N,512]x[512,1024] + bias/lrelu/max over N ----
// Weight o-tile (16 x 512 f16 = 16KB) staged in LDS with async-to-LDS copies;
// the 16 weight fragments are then hoisted into VGPRs (launch_bounds gives
// 2 waves/SIMD -> no spills). Inner loop: 2x global_load_b128 + wmma.
__global__ void __launch_bounds__(256, 1)
k_final(const _Float16* __restrict__ cat, const _Float16* __restrict__ wf,
        const float* __restrict__ bf, float* __restrict__ out) {
  int ot = blockIdx.x, b = blockIdx.y;
  int tid = threadIdx.x, lane = tid & 31, w = tid >> 5;
  int col = lane & 15, o0 = ot * 16;

  __shared__ _Float16 wl[16 * CCAT];                 // 16 KB weight tile
  {
    const _Float16* wsrc = wf + (size_t)o0 * CCAT;
    for (int i = tid; i < 16 * CCAT / 8; i += blockDim.x) {  // 16B per lane
      unsigned lo = (unsigned)(unsigned long long)&wl[i * 8];
      unsigned long long ga = (unsigned long long)(wsrc + i * 8);
      asm volatile("global_load_async_to_lds_b128 %0, %1, off"
                   :: "v"(lo), "v"(ga) : "memory");
    }
    asm volatile("s_wait_asynccnt 0x0" ::: "memory");
  }
  __syncthreads();

  const _Float16* cb = cat + (size_t)b * NPTS * CCAT;
  float bia = bf[o0 + col];
  float runmax = -3.4e38f;
  for (int mt = w; mt < NPTS / 16; mt += 8) {
    int n0 = mt * 16;
    if (mt + 8 < NPTS / 16)
      __builtin_prefetch(cb + (size_t)(n0 + 128 + (lane & 15)) * CCAT, 0, 1);
    v8f acc = {};
    #pragma unroll
    for (int kc = 0; kc < 16; ++kc) {
      v16h a  = load_frag(cb, n0, CCAT, kc * 32);
      v16h ww = load_frag(wl, 0, CCAT, kc * 32);     // LDS fragments (hoisted)
      acc = wmma32(a, ww, acc);
    }
    #pragma unroll
    for (int r = 0; r < 8; ++r) {
      float v = acc[r] + bia;
      v = v > 0.f ? v : 0.2f * v;
      runmax = fmaxf(runmax, v);
    }
  }
  __shared__ float red[256];
  red[tid] = runmax;
  __syncthreads();
  if (tid < 16) {               // tid%16 == lane%16 == output column
    float m = -3.4e38f;
    for (int t = tid; t < 256; t += 16) m = fmaxf(m, red[t]);
    out[(size_t)b * 1024 + o0 + tid] = m;
  }
}

// ---------------------------------------------------------------------------
extern "C" void kernel_launch(void* const* d_in, const int* in_sizes, int n_in,
                              void* d_out, int out_size, void* d_ws, size_t ws_size,
                              hipStream_t stream) {
  (void)in_sizes; (void)n_in; (void)out_size; (void)ws_size;
  const float* x  = (const float*)d_in[0];
  const float* Wl[4] = {(const float*)d_in[1], (const float*)d_in[3],
                        (const float*)d_in[5], (const float*)d_in[7]};
  const float* bl[4] = {(const float*)d_in[2], (const float*)d_in[4],
                        (const float*)d_in[6], (const float*)d_in[8]};
  const float* Wf = (const float*)d_in[9];
  const float* bf = (const float*)d_in[10];
  float* out = (float*)d_out;

  // workspace layout (~121 MB total; resident in 192MB L2)
  char* ws = (char*)d_ws;
  _Float16* h0   = (_Float16*)(ws);                       //  8,388,608  B*N*128 f16
  _Float16* h1   = (_Float16*)(ws + 8388608);             //  8,388,608
  float*    sqp  = (float*)   (ws + 16777216);            //    131,072  B*N f32
  int*      idxp = (int*)     (ws + 16908288);            //  2,621,440  B*N*20
  float*    abp  = (float*)   (ws + 19529728);            // 67,108,864  B*N*512 f32
  _Float16* catp = (_Float16*)(ws + 86638592);            // 33,554,432  B*N*512 f16
  _Float16* wbuf = (_Float16*)(ws + 120193024);           //  1,048,576  max(Wf) f16

  k_prep_x<<<(BATCH * NPTS) / 256, 256, 0, stream>>>(x, h0, sqp);

  // Layer 0: cin=3 (pad 32), O=64
  k_prep_w<<<128, 32, 0, stream>>>(Wl[0], wbuf, 3, 32, 64);
  k_knn<1><<<dim3(NPTS/16, BATCH), 32, 0, stream>>>(h0, sqp, idxp);
  k_gemm16<1, 4><<<dim3(NPTS/16, 2, BATCH), 32, 0, stream>>>(h0, wbuf, abp, 128);
  k_edge_max<<<dim3(NPTS, BATCH), 64, 0, stream>>>(abp, bl[0], idxp, h1, catp + 0, sqp, 64);

  // Layer 1: cin=64, O=64
  k_prep_w<<<128, 64, 0, stream>>>(Wl[1], wbuf, 64, 64, 64);
  k_knn<2><<<dim3(NPTS/16, BATCH), 32, 0, stream>>>(h1, sqp, idxp);
  k_gemm16<2, 4><<<dim3(NPTS/16, 2, BATCH), 32, 0, stream>>>(h1, wbuf, abp, 128);
  k_edge_max<<<dim3(NPTS, BATCH), 64, 0, stream>>>(abp, bl[1], idxp, h0, catp + 64, sqp, 64);

  // Layer 2: cin=64, O=128
  k_prep_w<<<256, 64, 0, stream>>>(Wl[2], wbuf, 64, 64, 128);
  k_knn<2><<<dim3(NPTS/16, BATCH), 32, 0, stream>>>(h0, sqp, idxp);
  k_gemm16<2, 4><<<dim3(NPTS/16, 4, BATCH), 32, 0, stream>>>(h0, wbuf, abp, 256);
  k_edge_max<<<dim3(NPTS, BATCH), 128, 0, stream>>>(abp, bl[2], idxp, h1, catp + 128, sqp, 128);

  // Layer 3: cin=128, O=256 (last edge layer: no next features / sq needed)
  k_prep_w<<<512, 128, 0, stream>>>(Wl[3], wbuf, 128, 128, 256);
  k_knn<4><<<dim3(NPTS/16, BATCH), 32, 0, stream>>>(h1, sqp, idxp);
  k_gemm16<4, 4><<<dim3(NPTS/16, 8, BATCH), 32, 0, stream>>>(h1, wbuf, abp, 512);
  k_edge_max<<<dim3(NPTS, BATCH), 256, 0, stream>>>(abp, bl[3], idxp, nullptr, catp + 256, nullptr, 256);

  // Final 512 -> 1024 + global max over points
  k_prep_wf<<<1024, 256, 0, stream>>>(Wf, wbuf);
  k_final<<<dim3(64, BATCH), 256, 0, stream>>>(catp, wbuf, bf, out);
}
